// FusedMoE_69501160784376
// MI455X (gfx1250) — matrix operand order repaired and checked by reading, not compiled, not run on hardware
//
#include <hip/hip_runtime.h>
#include <hip/hip_bf16.h>

#define NUM_EXPERTS 16
#define TOP_K 2
#define HIDDEN 1024
#define INTER 2048
#define NUM_TOKENS 4096
#define CAPACITY 768

typedef __bf16 bf16_t;
typedef bf16_t v16bf __attribute__((ext_vector_type(16)));
typedef bf16_t v8bf  __attribute__((ext_vector_type(8)));
typedef float  v8f   __attribute__((ext_vector_type(8)));
typedef float  v4f   __attribute__((ext_vector_type(4)));

// ---------------------------------------------------------------------------
// Router: softmax over 16 logits, top-2, renormalized scales.
// ---------------------------------------------------------------------------
__global__ void moe_router(const float* __restrict__ logits,
                           int* __restrict__ sel, float* __restrict__ scl) {
    int t = blockIdx.x * blockDim.x + threadIdx.x;
    if (t >= NUM_TOKENS) return;
    const float* l = logits + (size_t)t * NUM_EXPERTS;
    float v[NUM_EXPERTS];
    float mx = -1e30f;
    #pragma unroll
    for (int i = 0; i < NUM_EXPERTS; ++i) { v[i] = l[i]; mx = fmaxf(mx, v[i]); }
    float sum = 0.f;
    #pragma unroll
    for (int i = 0; i < NUM_EXPERTS; ++i) { v[i] = __expf(v[i] - mx); sum += v[i]; }
    int i1 = 0; float p1 = v[0];
    #pragma unroll
    for (int i = 1; i < NUM_EXPERTS; ++i) if (v[i] > p1) { p1 = v[i]; i1 = i; }
    int i2 = (i1 == 0) ? 1 : 0; float p2 = v[i2];
    #pragma unroll
    for (int i = 0; i < NUM_EXPERTS; ++i)
        if (i != i1 && v[i] > p2) { p2 = v[i]; i2 = i; }
    p1 /= sum; p2 /= sum;
    float d = p1 + p2;
    sel[2 * t]     = i1;
    sel[2 * t + 1] = i2;
    scl[2 * t]     = p1 / d;
    scl[2 * t + 1] = p2 / d;
}

// ---------------------------------------------------------------------------
// Zero the f32 output (each token receives exactly 2 atomic contributions).
// ---------------------------------------------------------------------------
__global__ void moe_zero_out(float* __restrict__ out) {
    size_t i = ((size_t)blockIdx.x * blockDim.x + threadIdx.x) * 4;
    v4f z = {0.f, 0.f, 0.f, 0.f};
    *(v4f*)(out + i) = z;
}

// ---------------------------------------------------------------------------
// Deterministic per-expert slot assignment (matches reference cumsum order).
// One wave32 per expert; ballot prefix-scan over flat (token,k) slots.
// ---------------------------------------------------------------------------
__global__ void moe_assign(const int* __restrict__ sel, const float* __restrict__ scl,
                           int* __restrict__ etok, float* __restrict__ escl,
                           int* __restrict__ cnt) {
    int e = blockIdx.x;
    int lane = threadIdx.x;
    for (int c = lane; c < CAPACITY; c += 32) {
        etok[e * CAPACITY + c] = 0;
        escl[e * CAPACITY + c] = 0.f;
    }
    __syncthreads();
    int n = 0;
    for (int base = 0; base < NUM_TOKENS * TOP_K; base += 32) {
        int s = base + lane;
        bool p = (sel[s] == e);
        unsigned mask = (unsigned)__ballot(p);
        int pos = n + __popc(mask & ((1u << lane) - 1u));
        if (p && pos < CAPACITY) {
            etok[e * CAPACITY + pos] = s;      // flat slot id
            escl[e * CAPACITY + pos] = scl[s];
        }
        n += __popc(mask);
    }
    if (lane == 0) cnt[e] = (n < CAPACITY) ? n : CAPACITY;
}

// ---------------------------------------------------------------------------
// Gather token rows into per-expert bf16 buffers; zero unused capacity rows.
// ---------------------------------------------------------------------------
__global__ void moe_gather(const float* __restrict__ hidden, const int* __restrict__ etok,
                           const int* __restrict__ cnt, bf16_t* __restrict__ buf) {
    int row = blockIdx.x;                     // e * CAPACITY + c
    int e = row / CAPACITY, c = row % CAPACITY;
    bf16_t* dst = buf + (size_t)row * HIDDEN + threadIdx.x * 8;
    v8bf o;
    if (c < cnt[e]) {
        int token = etok[row] / TOP_K;
        const float* src = hidden + (size_t)token * HIDDEN + threadIdx.x * 8;
        v4f a = *(const v4f*)src;
        v4f b = *(const v4f*)(src + 4);
        #pragma unroll
        for (int i = 0; i < 4; ++i) { o[i] = (bf16_t)a[i]; o[4 + i] = (bf16_t)b[i]; }
    } else {
        #pragma unroll
        for (int i = 0; i < 8; ++i) o[i] = (bf16_t)0.f;
    }
    *(v8bf*)dst = o;
}

// ---------------------------------------------------------------------------
// WMMA fragment loaders (16-bit, wave32; contiguous k order shared by A and B).
// A: 16x32 bf16, row-major source. Lane L: M=L&15, K = (L>>4)*16 .. +15.
// ---------------------------------------------------------------------------
__device__ __forceinline__ v16bf load_a_bf16(const bf16_t* __restrict__ A, int lda,
                                             int k0, int lane) {
    int m  = lane & 15;
    int ko = (lane >> 4) * 16;
    const bf16_t* p = A + (size_t)m * lda + k0 + ko;
    v8bf lo = *(const v8bf*)p;          // K = k0+ko      .. +7
    v8bf hi = *(const v8bf*)(p + 8);    // K = k0+8+ko    .. +7
    v16bf r;
    #pragma unroll
    for (int i = 0; i < 8; ++i) { r[i] = lo[i]; r[i + 8] = hi[i]; }
    return r;
}

// B: 32x16 bf16 from row-major (N,K) f32 weights. Lane L: N=L&15,
// K = (L>>4)*16 .. +15 -> 16 contiguous f32 per lane, converted to bf16.
__device__ __forceinline__ v16bf load_b_f32(const float* __restrict__ W, int ldk,
                                            int k0, int lane) {
    int n  = lane & 15;
    int ko = (lane >> 4) * 16;
    const float* p = W + (size_t)n * ldk + k0 + ko;
    v16bf r;
    #pragma unroll
    for (int i = 0; i < 16; i += 4) {
        v4f f = *(const v4f*)(p + i);
        r[i] = (bf16_t)f[0]; r[i+1] = (bf16_t)f[1];
        r[i+2] = (bf16_t)f[2]; r[i+3] = (bf16_t)f[3];
    }
    return r;
}

#define WMMA_BF16(a, b, c) \
    __builtin_amdgcn_wmma_f32_16x16x32_bf16(false, (a), false, (b), (short)0, (c), false, false)

// ---------------------------------------------------------------------------
// GEMM1: h = buf(C,H) x w3_w1(2I,H)^T, fused SiLU(g1)*g3 -> inter bf16 (C,I).
// 8 waves/WG; wave tile = 32M x 64N(I), paired w3/w1 columns.
// Per k-step: 2 A frags + 8 B frags -> 16 WMMAs (4 cvt-frag loads per WMMA pair).
// ---------------------------------------------------------------------------
#define MB1 4
#define NB1 4   // 4 * 512 = INTER
__global__ __launch_bounds__(256) void moe_gemm1(const bf16_t* __restrict__ buf,
                                                 const float* __restrict__ w31,
                                                 const int* __restrict__ cnt,
                                                 bf16_t* __restrict__ inter) {
    int wg = blockIdx.x;
    int e  = wg / (NB1 * MB1);
    int r0 = wg % (NB1 * MB1);
    int nb = r0 / MB1, mb = r0 % MB1;
    int wave = threadIdx.x >> 5, lane = threadIdx.x & 31;
    int n0 = nb * 512 + wave * 64;                    // column base in I-space
    const bf16_t* A = buf  + (size_t)e * CAPACITY * HIDDEN;
    const float*  W = w31  + (size_t)e * (2 * INTER) * HIDDEN;
    bf16_t*       O = inter + (size_t)e * CAPACITY * INTER;
    const int MT = (CAPACITY / 32) / MB1;             // 6 m-tile-pairs per WG
    int cmax = cnt[e];
    const v8f vzero = {0.f,0.f,0.f,0.f,0.f,0.f,0.f,0.f};
    int nl = lane & 15, mi = (lane >> 4) * 8;

    for (int mt = 0; mt < MT; ++mt) {
        int m0 = (mb * MT + mt) * 32;                 // pair of 16-row tiles
        if (m0 >= cmax) {
            // rows past capacity count: inter must still be well-defined zeros
            #pragma unroll
            for (int t = 0; t < 2; ++t)
                #pragma unroll
                for (int j = 0; j < 4; ++j)
                    #pragma unroll
                    for (int r = 0; r < 8; ++r)
                        O[(size_t)(m0 + t * 16 + mi + r) * INTER + (n0 + j * 16 + nl)] = (bf16_t)0.f;
            continue;
        }
        v8f acc3[2][4] = {{vzero, vzero, vzero, vzero}, {vzero, vzero, vzero, vzero}};
        v8f acc1[2][4] = {{vzero, vzero, vzero, vzero}, {vzero, vzero, vzero, vzero}};
        const bf16_t* Am0 = A + (size_t)m0 * HIDDEN;
        const bf16_t* Am1 = A + (size_t)(m0 + 16) * HIDDEN;
        for (int k0 = 0; k0 < HIDDEN; k0 += 32) {
            __builtin_prefetch((const void*)(W + (size_t)n0 * HIDDEN + k0 + 256), 0, 1);
            v16bf a0 = load_a_bf16(Am0, HIDDEN, k0, lane);
            v16bf a1 = load_a_bf16(Am1, HIDDEN, k0, lane);
            #pragma unroll
            for (int j = 0; j < 4; ++j) {
                v16bf b3 = load_b_f32(W + (size_t)(n0 + j * 16) * HIDDEN, HIDDEN, k0, lane);
                acc3[0][j] = WMMA_BF16(a0, b3, acc3[0][j]);
                acc3[1][j] = WMMA_BF16(a1, b3, acc3[1][j]);
                v16bf b1 = load_b_f32(W + (size_t)(INTER + n0 + j * 16) * HIDDEN, HIDDEN, k0, lane);
                acc1[0][j] = WMMA_BF16(a0, b1, acc1[0][j]);
                acc1[1][j] = WMMA_BF16(a1, b1, acc1[1][j]);
            }
        }
        #pragma unroll
        for (int t = 0; t < 2; ++t)
            #pragma unroll
            for (int j = 0; j < 4; ++j)
                #pragma unroll
                for (int r = 0; r < 8; ++r) {
                    float g1 = acc1[t][j][r];
                    float g3 = acc3[t][j][r];
                    float s  = g1 * __frcp_rn(1.f + __expf(-g1));   // SiLU(g1)
                    O[(size_t)(m0 + t * 16 + mi + r) * INTER + (n0 + j * 16 + nl)] =
                        (bf16_t)(s * g3);
                }
    }
}

// ---------------------------------------------------------------------------
// GEMM2: y = inter(C,I) x w2(H,I)^T ; fused scale + atomic scatter to out.
// Wave tile = 32M x 64N(H); per k-step: 2 A frags + 4 B frags -> 8 WMMAs.
// ---------------------------------------------------------------------------
#define MB2 4
#define NB2 2   // 2 * 512 = HIDDEN
__global__ __launch_bounds__(256) void moe_gemm2(const bf16_t* __restrict__ inter,
                                                 const float* __restrict__ w2,
                                                 const int* __restrict__ etok,
                                                 const float* __restrict__ escl,
                                                 const int* __restrict__ cnt,
                                                 float* __restrict__ out) {
    int wg = blockIdx.x;
    int e  = wg / (NB2 * MB2);
    int r0 = wg % (NB2 * MB2);
    int nb = r0 / MB2, mb = r0 % MB2;
    int wave = threadIdx.x >> 5, lane = threadIdx.x & 31;
    int n0 = nb * 512 + wave * 64;                    // column base in H-space
    const bf16_t* A = inter + (size_t)e * CAPACITY * INTER;
    const float*  W = w2    + (size_t)e * HIDDEN * INTER;
    const int MT = (CAPACITY / 32) / MB2;             // 6 m-tile-pairs per WG
    int cmax = cnt[e];
    const v8f vzero = {0.f,0.f,0.f,0.f,0.f,0.f,0.f,0.f};
    int nl = lane & 15, mi = (lane >> 4) * 8;

    for (int mt = 0; mt < MT; ++mt) {
        int m0 = (mb * MT + mt) * 32;
        if (m0 >= cmax) continue;                     // no output rows here
        v8f acc[2][4] = {{vzero, vzero, vzero, vzero}, {vzero, vzero, vzero, vzero}};
        const bf16_t* Am0 = A + (size_t)m0 * INTER;
        const bf16_t* Am1 = A + (size_t)(m0 + 16) * INTER;
        for (int k0 = 0; k0 < INTER; k0 += 32) {
            __builtin_prefetch((const void*)(W + (size_t)n0 * INTER + k0 + 256), 0, 1);
            v16bf a0 = load_a_bf16(Am0, INTER, k0, lane);
            v16bf a1 = load_a_bf16(Am1, INTER, k0, lane);
            #pragma unroll
            for (int j = 0; j < 4; ++j) {
                v16bf b = load_b_f32(W + (size_t)(n0 + j * 16) * INTER, INTER, k0, lane);
                acc[0][j] = WMMA_BF16(a0, b, acc[0][j]);
                acc[1][j] = WMMA_BF16(a1, b, acc[1][j]);
            }
        }
        #pragma unroll
        for (int t = 0; t < 2; ++t)
            #pragma unroll
            for (int j = 0; j < 4; ++j)
                #pragma unroll
                for (int r = 0; r < 8; ++r) {
                    int m = m0 + t * 16 + mi + r;
                    if (m < cmax) {
                        int   slot  = etok[e * CAPACITY + m];
                        float sc    = escl[e * CAPACITY + m];
                        int   token = slot / TOP_K;
                        atomicAdd(out + (size_t)token * HIDDEN + (n0 + j * 16 + nl),
                                  sc * acc[t][j][r]);
                    }
                }
    }
}

// ---------------------------------------------------------------------------
// Workspace layout (bytes):
//   sel   : 0          .. 32768        (8192 i32)
//   scl   : 32768      .. 65536        (8192 f32)
//   etok  : 65536      .. 114688       (16*768 i32)
//   escl  : 114688     .. 163840       (16*768 f32)
//   cnt   : 163840     .. 164096       (16 i32, padded)
//   buf   : 164096     .. +24 MiB      (16*768*1024 bf16)
//   inter : buf_end    .. +48 MiB      (16*768*2048 bf16)
// Total ~72.2 MB.
// ---------------------------------------------------------------------------
extern "C" void kernel_launch(void* const* d_in, const int* in_sizes, int n_in,
                              void* d_out, int out_size, void* d_ws, size_t ws_size,
                              hipStream_t stream) {
    const float* hidden = (const float*)d_in[0];
    const float* logits = (const float*)d_in[1];
    const float* w31    = (const float*)d_in[2];
    const float* w2     = (const float*)d_in[3];
    float* out = (float*)d_out;

    char* ws = (char*)d_ws;
    int*    sel   = (int*)  (ws + 0);
    float*  scl   = (float*)(ws + 32768);
    int*    etok  = (int*)  (ws + 65536);
    float*  escl  = (float*)(ws + 114688);
    int*    cnt   = (int*)  (ws + 163840);
    bf16_t* buf   = (bf16_t*)(ws + 164096);
    bf16_t* inter = (bf16_t*)(ws + 164096 + (size_t)NUM_EXPERTS * CAPACITY * HIDDEN * 2);

    moe_router  <<<(NUM_TOKENS + 255) / 256, 256, 0, stream>>>(logits, sel, scl);
    moe_zero_out<<<(NUM_TOKENS * HIDDEN) / (256 * 4), 256, 0, stream>>>(out);
    moe_assign  <<<NUM_EXPERTS, 32, 0, stream>>>(sel, scl, etok, escl, cnt);
    moe_gather  <<<NUM_EXPERTS * CAPACITY, 128, 0, stream>>>(hidden, etok, cnt, buf);
    moe_gemm1   <<<NUM_EXPERTS * NB1 * MB1, 256, 0, stream>>>(buf, w31, cnt, inter);
    moe_gemm2   <<<NUM_EXPERTS * NB2 * MB2, 256, 0, stream>>>(inter, w2, etok, escl, cnt, out);
}
